// Quant_Linear_7971459301616
// MI455X (gfx1250) — 8-bit matrix operand order repaired and checked by reading, compiled, not run on hardware
//
#include <hip/hip_runtime.h>
#include <stdint.h>

typedef __attribute__((ext_vector_type(8))) int v8i;

#define N_ROWS  65536
#define K_DIM   512
#define OUT_DIM 512
#define EPSQ    1e-8f

#define TM 128
#define TN 128
#define TK 128
#define SA 136
#define SB 144

__global__ __launch_bounds__(256)
void quant_weight_kernel(const float* __restrict__ weight,
                         const float* __restrict__ bias,
                         const float* __restrict__ scale_x,
                         int8_t* __restrict__ w_q8,
                         int*    __restrict__ b_q,
                         float*  __restrict__ scale_out)
{
    __shared__ float red[256];
    __shared__ float s_scale;

    const int o = blockIdx.x;
    const int t = threadIdx.x;
    const float* wrow = weight + (size_t)o * K_DIM;

    float a = fmaxf(fabsf(wrow[t]), fabsf(wrow[t + 256]));
    red[t] = a;
    __syncthreads();
    for (int s = 128; s > 0; s >>= 1) {
        if (t < s) red[t] = fmaxf(red[t], red[t + s]);
        __syncthreads();
    }
    if (t == 0) s_scale = fmaxf(red[0], EPSQ) * (1.0f / 127.0f);
    __syncthreads();

    const float inv = 1.0f / s_scale;
    for (int i = t; i < K_DIM; i += 256) {
        float q = rintf(wrow[i] * inv);
        q = fminf(fmaxf(q, -128.0f), 127.0f);
        w_q8[(size_t)o * K_DIM + i] = (int8_t)(int)q;
    }

    if (t == 0) {
        const float so = s_scale * scale_x[0];
        scale_out[o] = so;
        double bq = rint((double)bias[o] / (double)so);
        bq = fmin(fmax(bq, -2147483648.0), 2147483647.0);
        b_q[o] = (int)(long long)bq;
    }
}

__global__ __launch_bounds__(256)
void qgemm_wmma_kernel(const float*  __restrict__ x,
                       const int8_t* __restrict__ w_q8,
                       const int*    __restrict__ b_q,
                       int*          __restrict__ out)
{
    __shared__ __align__(16) unsigned char ldsA[TM * SA];
    __shared__ __align__(16) unsigned char ldsB[TN * SB];

    const int t    = threadIdx.x;
    const int lane = t & 31;
    const int wid  = t >> 5;

    const int rowBase = blockIdx.y * TM;
    const int colBase = blockIdx.x * TN;

    const int wm = (wid & 3) * 32;
    const int wn = (wid >> 2) * 64;

    const v8i vzero = {0, 0, 0, 0, 0, 0, 0, 0};
    v8i acc[2][4];
#pragma unroll
    for (int i = 0; i < 2; ++i)
#pragma unroll
        for (int j = 0; j < 4; ++j)
            acc[i][j] = vzero;

    const int a_row0 = (t >> 5) * 16;
    const int a_c4   = (t & 31);
    const int b_row0 = (t >> 3);
    const int b_k16  = (t & 7) * 16;

    for (int kc = 0; kc < K_DIM; kc += TK) {
        __syncthreads();

#pragma unroll
        for (int i = 0; i < 16; ++i) {
            const int r = a_row0 + i;
            const float4 f = *(const float4*)(x + (size_t)(rowBase + r) * K_DIM
                                                + kc + a_c4 * 4);
            const int q0 = __float2int_rn(f.x);
            const int q1 = __float2int_rn(f.y);
            const int q2 = __float2int_rn(f.z);
            const int q3 = __float2int_rn(f.w);
            const uint32_t packed = (uint32_t)(q0 & 0xFF)
                                  | ((uint32_t)(q1 & 0xFF) << 8)
                                  | ((uint32_t)(q2 & 0xFF) << 16)
                                  | ((uint32_t)(q3 & 0xFF) << 24);
            *(uint32_t*)(ldsA + r * SA + a_c4 * 4) = packed;
        }

#pragma unroll
        for (int i = 0; i < 4; ++i) {
            const int r = b_row0 + i * 32;
            const int4 v = *(const int4*)(w_q8 + (size_t)(colBase + r) * K_DIM
                                               + kc + b_k16);
            *(int4*)(ldsB + r * SB + b_k16) = v;
        }

        if (kc + TK < K_DIM) {
            __builtin_prefetch(x + (size_t)(rowBase + a_row0) * K_DIM
                                 + (kc + TK) + a_c4 * 4, 0, 1);
        }

        __syncthreads();

#pragma unroll
        for (int ks = 0; ks < 2; ++ks) {
            const int kb = ks * 64;

            v8i afrag[2];
#pragma unroll
            for (int mt = 0; mt < 2; ++mt) {
                const unsigned char* ab =
                    ldsA + (wm + mt * 16 + (lane & 15)) * SA + kb + (lane >> 4) * 16;
                const int2 d0 = *(const int2*)(ab + 0);
                const int2 d1 = *(const int2*)(ab + 8);
                const int2 d2 = *(const int2*)(ab + 32);
                const int2 d3 = *(const int2*)(ab + 40);
                v8i a;
                a[0] = d0.x; a[1] = d0.y; a[2] = d1.x; a[3] = d1.y;
                a[4] = d2.x; a[5] = d2.y; a[6] = d3.x; a[7] = d3.y;
                afrag[mt] = a;
            }

#pragma unroll
            for (int nt = 0; nt < 4; ++nt) {
                const unsigned char* bb =
                    ldsB + (wn + nt * 16 + (lane & 15)) * SB + kb + (lane >> 4) * 16;
                const int4 e0 = *(const int4*)(bb + 0);
                const int4 e1 = *(const int4*)(bb + 32);
                v8i b;
                b[0] = e0.x; b[1] = e0.y; b[2] = e0.z; b[3] = e0.w;
                b[4] = e1.x; b[5] = e1.y; b[6] = e1.z; b[7] = e1.w;
#pragma unroll
                for (int mt = 0; mt < 2; ++mt) {
                    acc[mt][nt] = __builtin_amdgcn_wmma_i32_16x16x64_iu8(
                        true, afrag[mt], true, b, acc[mt][nt], false, false);
                }
            }
        }
    }

#pragma unroll
    for (int nt = 0; nt < 4; ++nt) {
        const int col  = colBase + wn + nt * 16 + (lane & 15);
        const int bqv  = b_q[col];
#pragma unroll
        for (int mt = 0; mt < 2; ++mt) {
#pragma unroll
            for (int v = 0; v < 8; ++v) {
                const int row = rowBase + wm + mt * 16 + (lane >> 4) * 8 + v;
                out[(size_t)row * OUT_DIM + col] = acc[mt][nt][v] + bqv;
            }
        }
    }
}

extern "C" void kernel_launch(void* const* d_in, const int* in_sizes, int n_in,
                              void* d_out, int out_size, void* d_ws, size_t ws_size,
                              hipStream_t stream)
{
    const float* x_q     = (const float*)d_in[0];
    const float* weight  = (const float*)d_in[1];
    const float* bias    = (const float*)d_in[2];
    const float* scale_x = (const float*)d_in[3];

    int8_t* w_q8 = (int8_t*)d_ws;
    int*    b_q  = (int*)((char*)d_ws + (size_t)OUT_DIM * K_DIM);

    int*   out_q     = (int*)d_out;
    float* scale_out = (float*)d_out + (out_size - OUT_DIM);

    quant_weight_kernel<<<OUT_DIM, 256, 0, stream>>>(weight, bias, scale_x,
                                                     w_q8, b_q, scale_out);

    dim3 grid(OUT_DIM / TN, N_ROWS / TM);
    qgemm_wmma_kernel<<<grid, 256, 0, stream>>>(x_q, w_q8, b_q, out_q);
}
